// FullSelfAttention_10952166605227
// MI455X (gfx1250) — matrix operand order repaired and checked by reading, not compiled, not run on hardware
//
#include <hip/hip_runtime.h>
#include <hip/hip_bf16.h>
#include <math.h>

typedef __attribute__((ext_vector_type(16))) _Float16 v16h;
typedef __attribute__((ext_vector_type(8)))  _Float16 v8h;
typedef __attribute__((ext_vector_type(4)))  _Float16 v4h;
typedef __attribute__((ext_vector_type(8)))  float    v8f;

#define BATCH 2
#define SEQ   2048
#define DIM   1024
#define HEADS 16
#define HDIM  64
#define BT    (BATCH * SEQ)     // 4096 rows
#define QKV_N (3 * DIM)         // 3072

// A-fragment (16-bit, 16x32) K-mapping per CDNA5 ISA 7.12.2 reduces to two
// contiguous 8-half runs: j=0..7 -> k = hgrp*8 + j ; j=8..15 -> k = 16+hgrp*8+(j-8)
__device__ __forceinline__ int a_k_of(int j, int hgrp) {
  int vg = j >> 1, sub = j & 1;
  return (vg < 4) ? (vg * 2 + sub + hgrp * 8)
                  : (16 + (vg - 4) * 2 + sub + hgrp * 8);
}

__device__ __forceinline__ v16h cat8(v8h lo, v8h hi) {
  v16h r;
#pragma unroll
  for (int j = 0; j < 8; ++j) { r[j] = lo[j]; r[j + 8] = hi[j]; }
  return r;
}

__device__ __forceinline__ v16h load_afrag(const _Float16* row, int k0, int hgrp) {
  v8h lo = *(const v8h*)(row + k0 + hgrp * 16);
  v8h hi = *(const v8h*)(row + k0 + hgrp * 16 + 8);
  return cat8(lo, hi);
}

__device__ __forceinline__ v8f wmma_f16(v16h a, v16h b, v8f c) {
  return __builtin_amdgcn_wmma_f32_16x16x32_f16(false, a, false, b,
                                                (short)0, c, false, false);
}

// ---------------------------------------------------------------------------
// Kernel 0: f32 -> f16 conversion, 4 elements/thread (b128 in, b64 out)
// ---------------------------------------------------------------------------
__global__ void cvt_f16_kernel(const float* __restrict__ src,
                               _Float16* __restrict__ dst, int n4) {
  int i = blockIdx.x * blockDim.x + threadIdx.x;
  if (i >= n4) return;
  float4 v = ((const float4*)src)[i];
  v4h o;
  o[0] = (_Float16)v.x; o[1] = (_Float16)v.y;
  o[2] = (_Float16)v.z; o[3] = (_Float16)v.w;
  ((v4h*)dst)[i] = o;
}

// ---------------------------------------------------------------------------
// Kernel 1: Out(f32, MxN) = A(f16, MxK row-major) @ W(f16, NxK row-major)^T
// K = DIM. Block = 128 threads = 4 waves; 32(M) x 64(N) tile per wave:
// 2 A-frags + 4 B-frags -> 8 WMMAs per 32-K step, pure f16 vector loads.
// ---------------------------------------------------------------------------
__global__ void gemm_f16_kernel(const _Float16* __restrict__ A,
                                const _Float16* __restrict__ W,
                                float* __restrict__ Out, int N) {
  const int K = DIM;
  int lane = threadIdx.x & 31;
  int wv   = threadIdx.x >> 5;
  int n0   = (blockIdx.x * 4 + wv) * 64;
  int m0   = blockIdx.y * 32;
  int nm   = lane & 15;
  int hgrp = lane >> 4;
  const _Float16* ar0 = A + (size_t)(m0 + nm) * K;
  const _Float16* ar1 = A + (size_t)(m0 + 16 + nm) * K;
  const _Float16* wr[4];
#pragma unroll
  for (int i = 0; i < 4; ++i) wr[i] = W + (size_t)(n0 + i * 16 + nm) * K;

  v8f c[2][4] = {};
  for (int k0 = 0; k0 < K; k0 += 32) {
    v16h a0 = load_afrag(ar0, k0, hgrp);
    v16h a1 = load_afrag(ar1, k0, hgrp);
#pragma unroll
    for (int i = 0; i < 4; ++i) {
      v16h b = *(const v16h*)(wr[i] + k0 + hgrp * 16);  // linear-K B fragment
      c[0][i] = wmma_f16(a0, b, c[0][i]);
      c[1][i] = wmma_f16(a1, b, c[1][i]);
    }
  }
#pragma unroll
  for (int mt = 0; mt < 2; ++mt)
#pragma unroll
    for (int i = 0; i < 4; ++i)
#pragma unroll
      for (int r = 0; r < 8; ++r)
        Out[(size_t)(m0 + mt * 16 + r + 8 * hgrp) * N + n0 + i * 16 + nm] =
            c[mt][i][r];
}

// ---------------------------------------------------------------------------
// Kernel 2: per (b,t,h): RMS-norm q,k over 64 dims; RoPE; emit f16 q/k/v in
// (B,H,T,D) layout. One wave per (b,t,h); lane j owns dims j and j+32 (the
// RoPE x1/x2 pair).
// ---------------------------------------------------------------------------
__global__ void qkv_post_kernel(const float* __restrict__ QKV,
                                _Float16* __restrict__ Q16,
                                _Float16* __restrict__ K16,
                                _Float16* __restrict__ V16) {
  int bt = blockIdx.x;               // 0..BT-1
  int h  = blockIdx.y;               // 0..15
  int t  = bt & (SEQ - 1);
  int b  = bt >> 11;                 // SEQ = 2^11
  int j  = threadIdx.x;              // 0..31
  const float* row = QKV + (size_t)bt * QKV_N + h * (3 * HDIM);
  float q1 = row[j],       q2 = row[j + 32];
  float k1 = row[64 + j],  k2 = row[64 + j + 32];
  float v1 = row[128 + j], v2 = row[128 + j + 32];

  float sq = q1 * q1 + q2 * q2;
  float sk = k1 * k1 + k2 * k2;
#pragma unroll
  for (int off = 16; off >= 1; off >>= 1) {
    sq += __shfl_xor(sq, off, 32);
    sk += __shfl_xor(sk, off, 32);
  }
  float invq = rsqrtf(sq * (1.0f / 64.0f) + 1e-6f);
  float invk = rsqrtf(sk * (1.0f / 64.0f) + 1e-6f);
  q1 *= invq; q2 *= invq; k1 *= invk; k2 *= invk;

  // angular_freq[j] = (1/1024)^(j/15) for j<16, else 0
  float freq = (j < 16) ? powf(1024.0f, -(float)j * (1.0f / 15.0f)) : 0.0f;
  float th = (float)t * freq;
  float c = cosf(th), s = sinf(th);
  float qy1 =  q1 * c + q2 * s, qy2 = -q1 * s + q2 * c;
  float ky1 =  k1 * c + k2 * s, ky2 = -k1 * s + k2 * c;

  size_t base = ((size_t)(b * HEADS + h) * SEQ + t) * HDIM;
  Q16[base + j] = (_Float16)qy1; Q16[base + 32 + j] = (_Float16)qy2;
  K16[base + j] = (_Float16)ky1; K16[base + 32 + j] = (_Float16)ky2;
  V16[base + j] = (_Float16)v1;  V16[base + 32 + j] = (_Float16)v2;
}

// ---------------------------------------------------------------------------
// Kernel 2b: V (B,H,T,D) -> VT (B,H,D,T), LDS-tiled 64x64 transpose,
// coalesced b128 in and out. Block = 256 threads, one 64(T)x64(D) tile.
// ---------------------------------------------------------------------------
__global__ void vt_kernel(const _Float16* __restrict__ V16,
                          _Float16* __restrict__ VT) {
  __shared__ _Float16 tile[64][72];      // +8 pad: dodge bank conflicts
  int bh = blockIdx.y;
  int t0 = blockIdx.x * 64;
  int tid = threadIdx.x;
  int r  = tid >> 3;                     // 0..31
  int c8 = (tid & 7) * 8;                // 0,8,...,56
  const _Float16* src = V16 + (size_t)bh * SEQ * HDIM;
#pragma unroll
  for (int i = 0; i < 2; ++i) {
    int row = r + 32 * i;                // t within tile
    v8h v = *(const v8h*)(src + (size_t)(t0 + row) * HDIM + c8);
#pragma unroll
    for (int j = 0; j < 8; ++j) tile[c8 + j][row] = v[j];
  }
  __syncthreads();
  _Float16* dstb = VT + (size_t)bh * HDIM * SEQ;
#pragma unroll
  for (int i = 0; i < 2; ++i) {
    int d = r + 32 * i;                  // 0..63
    v8h v;
#pragma unroll
    for (int j = 0; j < 8; ++j) v[j] = tile[d][c8 + j];
    *(v8h*)(dstb + (size_t)d * SEQ + t0 + c8) = v;
  }
}

// ---------------------------------------------------------------------------
// Kernel 3: flash attention, one wave per (b,h, 32-query tile).
// Per 32-key iteration: 4 contiguous K-frag loads -> 8 score WMMAs, online
// softmax for both M-tiles, LDS transpose of P (32x32 f16), 4 contiguous
// V^T-frag loads -> 8 PV WMMAs.
// ---------------------------------------------------------------------------
__global__ void attn_kernel(const _Float16* __restrict__ Q16,
                            const _Float16* __restrict__ K16,
                            const _Float16* __restrict__ VT,
                            _Float16* __restrict__ Y16) {
  __shared__ _Float16 Plds[32 * 32];
  int lane = threadIdx.x;
  int nm   = lane & 15;
  int hgrp = lane >> 4;
  int bh = blockIdx.y;
  int b  = bh >> 4, h = bh & 15;
  int q0 = blockIdx.x * 32;
  const _Float16* Qb  = Q16 + (size_t)bh * SEQ * HDIM;
  const _Float16* Kb  = K16 + (size_t)bh * SEQ * HDIM;
  const _Float16* VTb = VT  + (size_t)bh * HDIM * SEQ;

  v16h qa[2][2];                    // [m-tile][d-chunk] A-fragments
#pragma unroll
  for (int mt = 0; mt < 2; ++mt) {
    const _Float16* qrow = Qb + (size_t)(q0 + mt * 16 + nm) * HDIM;
    qa[mt][0] = load_afrag(qrow, 0, hgrp);
    qa[mt][1] = load_afrag(qrow, 32, hgrp);
  }

  v8f o[2][4] = {};                 // [m-tile][d-tile] output accumulators
  float mst[2][8], lst[2][8];
#pragma unroll
  for (int mt = 0; mt < 2; ++mt)
#pragma unroll
    for (int r = 0; r < 8; ++r) { mst[mt][r] = -1e30f; lst[mt][r] = 0.0f; }
  const float scale = 0.125f;       // 1/sqrt(64)

  for (int k0 = 0; k0 < SEQ; k0 += 32) {
    // ---- scores: s[m-tile][key-tile], each K-frag feeds both m-tiles ----
    v8f s[2][2] = {};
#pragma unroll
    for (int kt = 0; kt < 2; ++kt) {
      const _Float16* kr = Kb + (size_t)(k0 + kt * 16 + nm) * HDIM;
#pragma unroll
      for (int dc = 0; dc < 2; ++dc) {
        v16h kb = *(const v16h*)(kr + dc * 32 + hgrp * 16);
        s[0][kt] = wmma_f16(qa[0][dc], kb, s[0][kt]);
        s[1][kt] = wmma_f16(qa[1][dc], kb, s[1][kt]);
      }
    }

    __syncthreads();
    // ---- online softmax per m-tile, write P into LDS (C->A transpose) ----
#pragma unroll
    for (int mt = 0; mt < 2; ++mt) {
#pragma unroll
      for (int r = 0; r < 8; ++r) {
        float a0 = s[mt][0][r] * scale;
        float a1 = s[mt][1][r] * scale;
        float rmax = fmaxf(a0, a1);
#pragma unroll
        for (int off = 8; off >= 1; off >>= 1)
          rmax = fmaxf(rmax, __shfl_xor(rmax, off, 32));  // 16-lane group
        float mnew = fmaxf(mst[mt][r], rmax);
        float fac  = expf(mst[mt][r] - mnew);
        float p0   = expf(a0 - mnew);
        float p1   = expf(a1 - mnew);
        float rsum = p0 + p1;
#pragma unroll
        for (int off = 8; off >= 1; off >>= 1)
          rsum += __shfl_xor(rsum, off, 32);
        lst[mt][r] = lst[mt][r] * fac + rsum;
        mst[mt][r] = mnew;
#pragma unroll
        for (int i = 0; i < 4; ++i) o[mt][i][r] *= fac;
        int m = mt * 16 + r + 8 * hgrp;                   // C-layout row
        Plds[m * 32 + nm]      = (_Float16)p0;
        Plds[m * 32 + 16 + nm] = (_Float16)p1;
      }
    }
    __syncthreads();

    // ---- P*V: each V-frag feeds both m-tiles ----
    v16h pa[2];
#pragma unroll
    for (int mt = 0; mt < 2; ++mt)
      pa[mt] = load_afrag(Plds + (mt * 16 + nm) * 32, 0, hgrp);

#pragma unroll
    for (int i = 0; i < 4; ++i) {
      v16h vf = *(const v16h*)(VTb + (size_t)(i * 16 + nm) * SEQ + k0 + hgrp * 16);
      o[0][i] = wmma_f16(pa[0], vf, o[0][i]);
      o[1][i] = wmma_f16(pa[1], vf, o[1][i]);
    }
  }

  // normalize and emit y as f16 in (B, T, H*D) row-major for the proj GEMM
#pragma unroll
  for (int mt = 0; mt < 2; ++mt)
#pragma unroll
    for (int r = 0; r < 8; ++r) {
      float inv = 1.0f / lst[mt][r];
      int t = q0 + mt * 16 + r + 8 * hgrp;
      size_t rowb = (size_t)(b * SEQ + t) * DIM + h * HDIM;
      Y16[rowb + 0  + nm] = (_Float16)(o[mt][0][r] * inv);
      Y16[rowb + 16 + nm] = (_Float16)(o[mt][1][r] * inv);
      Y16[rowb + 32 + nm] = (_Float16)(o[mt][2][r] * inv);
      Y16[rowb + 48 + nm] = (_Float16)(o[mt][3][r] * inv);
    }
}

// ---------------------------------------------------------------------------
extern "C" void kernel_launch(void* const* d_in, const int* in_sizes, int n_in,
                              void* d_out, int out_size, void* d_ws, size_t ws_size,
                              hipStream_t stream) {
  const float* x      = (const float*)d_in[0];   // (B, T, C) fp32
  const float* qkv_w  = (const float*)d_in[1];   // (3C, C)  fp32
  const float* proj_w = (const float*)d_in[2];   // (C, C)   fp32
  float* out = (float*)d_out;                    // (B, T, C) fp32

  char* ws = (char*)d_ws;
  size_t off = 0;
  float*    qkv  = (float*)(ws + off);    off += (size_t)BT * QKV_N * sizeof(float);
  _Float16* q16  = (_Float16*)(ws + off); off += (size_t)BT * DIM * sizeof(_Float16);
  _Float16* k16  = (_Float16*)(ws + off); off += (size_t)BT * DIM * sizeof(_Float16);
  _Float16* v16  = (_Float16*)(ws + off); off += (size_t)BT * DIM * sizeof(_Float16);
  _Float16* vt16 = (_Float16*)(ws + off); off += (size_t)BT * DIM * sizeof(_Float16);
  _Float16* y16  = (_Float16*)(ws + off); off += (size_t)BT * DIM * sizeof(_Float16);
  _Float16* x16  = (_Float16*)(ws + off); off += (size_t)BT * DIM * sizeof(_Float16);
  _Float16* wq16 = (_Float16*)(ws + off); off += (size_t)QKV_N * DIM * sizeof(_Float16);
  _Float16* wp16 = (_Float16*)(ws + off); off += (size_t)DIM * DIM * sizeof(_Float16);

  // one-time f32 -> f16 conversions (activations + weights)
  cvt_f16_kernel<<<(BT * DIM / 4 + 255) / 256, 256, 0, stream>>>(x, x16, BT * DIM / 4);
  cvt_f16_kernel<<<(QKV_N * DIM / 4 + 255) / 256, 256, 0, stream>>>(qkv_w, wq16, QKV_N * DIM / 4);
  cvt_f16_kernel<<<(DIM * DIM / 4 + 255) / 256, 256, 0, stream>>>(proj_w, wp16, DIM * DIM / 4);

  // block tile: 32(M) x 256(N), 4 waves/block
  gemm_f16_kernel<<<dim3(QKV_N / 256, BT / 32), 128, 0, stream>>>(x16, wq16, qkv, QKV_N);
  qkv_post_kernel<<<dim3(BT, HEADS), 32, 0, stream>>>(qkv, q16, k16, v16);
  vt_kernel<<<dim3(SEQ / 64, BATCH * HEADS), 256, 0, stream>>>(v16, vt16);
  attn_kernel<<<dim3(SEQ / 32, BATCH * HEADS), 32, 0, stream>>>(q16, k16, vt16, y16);
  gemm_f16_kernel<<<dim3(DIM / 256, BT / 32), 128, 0, stream>>>(y16, wp16, out, DIM);
}
